// VibeVoiceRelativePositionalEncoding_52381421142362
// MI455X (gfx1250) — matrix operand order repaired and checked by reading, not compiled, not run on hardware
//
#include <hip/hip_runtime.h>

typedef __attribute__((ext_vector_type(16))) _Float16 v16h;
typedef __attribute__((ext_vector_type(4)))  _Float16 v4h;
typedef __attribute__((ext_vector_type(8)))  float    v8f;

#define S_LEN    2048
#define D_DIM    64
#define BH_NUM   32
#define NUM_EMB  8192
#define HALF_EMB 4096
#define U_TILES  129   // band coverage in skew space: ceil((S+15)/16)

// out[b,h,i,j] = scores[b,h,i,j] + sum_d Q[b,h,i,d] * E[j-i+4096, d]
// Skew u = j - i turns the Toeplitz gather into a plain GEMM per (b,h):
//   out_skew[i,u] = Q[i,:] . E[4096+u,:]
// One wave per 16x16 skew tile, K=64 via two v_wmma_f32_16x16x32_f16.

// Convert two float4 (8 consecutive K values) into 8 f16 entries of a fragment.
__device__ __forceinline__ void cvt8(v16h& dst, int base, float4 lo, float4 hi) {
    dst[base + 0] = (_Float16)lo.x;  dst[base + 1] = (_Float16)lo.y;
    dst[base + 2] = (_Float16)lo.z;  dst[base + 3] = (_Float16)lo.w;
    dst[base + 4] = (_Float16)hi.x;  dst[base + 5] = (_Float16)hi.y;
    dst[base + 6] = (_Float16)hi.z;  dst[base + 7] = (_Float16)hi.w;
}

// Prologue: rel_emb f32 -> f16 in workspace (1 MB; E rows are reused by
// thousands of tiles, so pay the conversion once, not per tile).
__global__ __launch_bounds__(256)
void cvt_emb_kernel(const float* __restrict__ e, _Float16* __restrict__ o) {
    const int idx = blockIdx.x * 256 + threadIdx.x;   // NUM_EMB*D/4 threads
    const float4 v = ((const float4*)e)[idx];
    v4h h;
    h[0] = (_Float16)v.x; h[1] = (_Float16)v.y;
    h[2] = (_Float16)v.z; h[3] = (_Float16)v.w;
    ((v4h*)o)[idx] = h;
}

template <bool EMB16>
__global__ __launch_bounds__(256)
void relpos_wmma_kernel(const float* __restrict__ query,
                        const float* __restrict__ scores,
                        const float* __restrict__ rel_emb,
                        const _Float16* __restrict__ emb_h,
                        float* __restrict__ out)
{
    const int bh   = blockIdx.x;        // b*H + h
    const int i0   = blockIdx.y << 4;   // i-tile base
    // readfirstlane -> wave id in SGPR: t-loop + edge test become scalar
    // branches and EXEC stays all-1s at every WMMA.
    const int wave = __builtin_amdgcn_readfirstlane(threadIdx.x >> 5);
    const int lane = threadIdx.x & 31;
    const int half = lane >> 4;
    const int m    = lane & 15;         // A: row M ; B/C/D: column N

    // ---- A fragments (Q tile, loaded once per wave): contiguous K order (same as B) ----
    // frag0 per-lane K = 16h+0..15; frag1 = +32.
    v16h a0, a1;
    {
        const float4* q = (const float4*)(query +
            ((size_t)bh * S_LEN + (size_t)(i0 + m)) * D_DIM);
        cvt8(a0, 0, q[4 * half + 0],      q[4 * half + 1]);
        cvt8(a0, 8, q[4 * half + 2],      q[4 * half + 3]);
        cvt8(a1, 0, q[8 + 4 * half + 0],  q[8 + 4 * half + 1]);
        cvt8(a1, 8, q[8 + 4 * half + 2],  q[8 + 4 * half + 3]);
    }

    const float* sc_base  = scores + (size_t)bh * S_LEN * S_LEN;
    float*       out_base = out    + (size_t)bh * S_LEN * S_LEN;

    const int rowb = i0 + 8 * half;          // first of this lane-half's 8 rows
    const int lin0 = rowb * (S_LEN + 1);     // irow*S + j = irow*(S+1) + (u0+m)

    for (int t = wave; t < U_TILES; t += 8) {
        const int u0 = -i0 - 15 + 16 * t;    // E row idx stays in [2049, 6144]

        // ---- B fragments: B[k][n] = E[4096+u0+n, k] ----
        // B K order chosen to match A: per lane-half K = 16h+0..15 (16 consecutive d's).
        v16h b0, b1;
        if constexpr (EMB16) {
            // pre-converted f16: 32 contiguous bytes -> two b128 per fragment
            const _Float16* eh = emb_h +
                (size_t)(HALF_EMB + u0 + m) * D_DIM + 16 * half;
            b0 = *(const v16h*)(eh);        // K = 16h + 0..15
            b1 = *(const v16h*)(eh + 32);   // K = 32 + 16h + 0..15
        } else {
            const float4* e = (const float4*)(rel_emb +
                (size_t)(HALF_EMB + u0 + m) * D_DIM);
            cvt8(b0, 0, e[4 * half + 0],     e[4 * half + 1]);
            cvt8(b0, 8, e[4 * half + 2],     e[4 * half + 3]);
            cvt8(b1, 0, e[8 + 4 * half + 0], e[8 + 4 * half + 1]);
            cvt8(b1, 8, e[8 + 4 * half + 2], e[8 + 4 * half + 3]);
        }

        v8f c;
        if (t >= 1 && t <= U_TILES - 2) {
            // ----- fast path: whole tile in-range (scalar-uniform test) -----
            // 8 NT loads / 8 NT stores at immediate offsets off one address.
            const float* sp = sc_base + (lin0 + u0 + m);
            float*       op = out_base + (lin0 + u0 + m);
#pragma unroll
            for (int r = 0; r < 8; ++r)
                c[r] = __builtin_nontemporal_load(sp + r * (S_LEN + 1));

            c = __builtin_amdgcn_wmma_f32_16x16x32_f16(false, a0, false, b0,
                                                       (short)0, c, false, false);
            c = __builtin_amdgcn_wmma_f32_16x16x32_f16(false, a1, false, b1,
                                                       (short)0, c, false, false);
#pragma unroll
            for (int r = 0; r < 8; ++r)
                __builtin_nontemporal_store(c[r], op + r * (S_LEN + 1));
        } else {
            // ----- edge path (t==0 or t==128 only) --------------------------
            // Clamped-address loads + select: full EXEC before the WMMA.
#pragma unroll
            for (int r = 0; r < 8; ++r) {
                const int j  = rowb + r + u0 + m;
                const int jc = j < 0 ? 0 : (j > S_LEN - 1 ? S_LEN - 1 : j);
                const float v = sc_base[(size_t)(rowb + r) * S_LEN + jc];
                c[r] = (j == jc) ? v : 0.0f;
            }
            c = __builtin_amdgcn_wmma_f32_16x16x32_f16(false, a0, false, b0,
                                                       (short)0, c, false, false);
            c = __builtin_amdgcn_wmma_f32_16x16x32_f16(false, a1, false, b1,
                                                       (short)0, c, false, false);
#pragma unroll
            for (int r = 0; r < 8; ++r) {
                const int j = rowb + r + u0 + m;
                if (j >= 0 && j < S_LEN)
                    out_base[(size_t)(rowb + r) * S_LEN + j] = c[r];
            }
        }
    }
}

extern "C" void kernel_launch(void* const* d_in, const int* in_sizes, int n_in,
                              void* d_out, int out_size, void* d_ws, size_t ws_size,
                              hipStream_t stream) {
    const float* query   = (const float*)d_in[0];  // [B,H,S,D] f32
    const float* scores  = (const float*)d_in[1];  // [B,H,S,S] f32
    const float* rel_emb = (const float*)d_in[2];  // [NUM_EMB,D] f32
    float*       out     = (float*)d_out;          // [B,H,S,S] f32

    const size_t emb16_bytes = (size_t)NUM_EMB * D_DIM * sizeof(_Float16); // 1 MB
    dim3 grid(BH_NUM, S_LEN / 16);
    dim3 block(256);

    if (ws_size >= emb16_bytes) {
        _Float16* emb_h = (_Float16*)d_ws;
        cvt_emb_kernel<<<NUM_EMB * D_DIM / 4 / 256, 256, 0, stream>>>(rel_emb, emb_h);
        relpos_wmma_kernel<true><<<grid, block, 0, stream>>>(
            query, scores, rel_emb, emb_h, out);
    } else {
        relpos_wmma_kernel<false><<<grid, block, 0, stream>>>(
            query, scores, rel_emb, nullptr, out);
    }
}
